// BitLinear_24962349924854
// MI455X (gfx1250) — compile-verified, matrix operand order checked by reading, not run on hardware
//
#include <hip/hip_runtime.h>
#include <cstdint>

typedef __attribute__((ext_vector_type(8))) int v8i;

#define KDIM 4096
#define NDIM 4096
#define MDIM 8192
#define EPSV 1e-5f

// ---------------------------------------------------------------------------
// Kernel 1: per-token activation quantization.
// ---------------------------------------------------------------------------
__global__ __launch_bounds__(256) void quant_act_kernel(
    const float* __restrict__ x, int8_t* __restrict__ q, float* __restrict__ recip_s) {
  const int token = blockIdx.x;
  const int t = threadIdx.x;
  const float* row = x + (size_t)token * KDIM;

  float4 v[4];
  float mx = 0.0f;
#pragma unroll
  for (int i = 0; i < 4; ++i) {
    v[i] = ((const float4*)row)[t * 4 + i];
    mx = fmaxf(mx, fabsf(v[i].x));
    mx = fmaxf(mx, fabsf(v[i].y));
    mx = fmaxf(mx, fabsf(v[i].z));
    mx = fmaxf(mx, fabsf(v[i].w));
  }

  __shared__ float red[256];
  red[t] = mx;
  __syncthreads();
#pragma unroll
  for (int s = 128; s > 0; s >>= 1) {
    if (t < s) red[t] = fmaxf(red[t], red[t + s]);
    __syncthreads();
  }
  const float amax = fmaxf(red[0], EPSV);
  const float s = 127.0f / amax;
  if (t == 0) recip_s[token] = amax / 127.0f;

  uint32_t packed[4];
#pragma unroll
  for (int i = 0; i < 4; ++i) {
    const float* f = (const float*)&v[i];
    uint32_t w = 0;
#pragma unroll
    for (int b = 0; b < 4; ++b) {
      int iv = (int)rintf(f[b] * s);
      iv = iv < -128 ? -128 : (iv > 127 ? 127 : iv);
      w |= ((uint32_t)(uint8_t)(int8_t)iv) << (8 * b);
    }
    packed[i] = w;
  }
  uint4 out;
  out.x = packed[0]; out.y = packed[1]; out.z = packed[2]; out.w = packed[3];
  ((uint4*)(q + (size_t)token * KDIM))[t] = out;
}

// ---------------------------------------------------------------------------
// Kernel 2a: per-row |w| partial sums (deterministic, atomics-free).
// ---------------------------------------------------------------------------
__global__ __launch_bounds__(256) void wabs_partial_kernel(
    const float* __restrict__ w, float* __restrict__ partial) {
  const int row = blockIdx.x;
  const int t = threadIdx.x;
  const float* r = w + (size_t)row * KDIM;
  float sum = 0.0f;
#pragma unroll
  for (int i = 0; i < 4; ++i) {
    float4 v = ((const float4*)r)[t * 4 + i];
    sum += fabsf(v.x) + fabsf(v.y) + fabsf(v.z) + fabsf(v.w);
  }
  __shared__ float red[256];
  red[t] = sum;
  __syncthreads();
#pragma unroll
  for (int s = 128; s > 0; s >>= 1) {
    if (t < s) red[t] += red[t + s];
    __syncthreads();
  }
  if (t == 0) partial[row] = red[0];
}

// ---------------------------------------------------------------------------
// Kernel 2b: final reduction -> scale_w and inv(scale_w)=clipped mean|w|.
// ---------------------------------------------------------------------------
__global__ __launch_bounds__(256) void wabs_final_kernel(
    const float* __restrict__ partial, float* __restrict__ scal) {
  const int t = threadIdx.x;
  float sum = 0.0f;
#pragma unroll
  for (int i = 0; i < 16; ++i) sum += partial[t * 16 + i];
  __shared__ float red[256];
  red[t] = sum;
  __syncthreads();
#pragma unroll
  for (int s = 128; s > 0; s >>= 1) {
    if (t < s) red[t] += red[t + s];
    __syncthreads();
  }
  if (t == 0) {
    float mean = red[0] / ((float)NDIM * (float)KDIM);
    mean = fmaxf(mean, EPSV);
    scal[0] = 1.0f / mean;  // scale_w
    scal[1] = mean;         // 1/scale_w (final dequant factor)
  }
}

// ---------------------------------------------------------------------------
// Kernel 3: ternarize weights -> int8 {-1,0,1}, 16 elems/thread.
// ---------------------------------------------------------------------------
__global__ __launch_bounds__(256) void ternarize_kernel(
    const float* __restrict__ w, const float* __restrict__ scal, int8_t* __restrict__ tw) {
  const float sw = scal[0];
  const size_t base = ((size_t)blockIdx.x * 256 + threadIdx.x) * 16;
  uint32_t packed[4];
#pragma unroll
  for (int i = 0; i < 4; ++i) {
    float4 v = *(const float4*)(w + base + i * 4);
    const float* f = (const float*)&v;
    uint32_t word = 0;
#pragma unroll
    for (int b = 0; b < 4; ++b) {
      int iv = (int)rintf(f[b] * sw);
      iv = iv < -1 ? -1 : (iv > 1 ? 1 : iv);
      word |= ((uint32_t)(uint8_t)(int8_t)iv) << (8 * b);
    }
    packed[i] = word;
  }
  uint4 o;
  o.x = packed[0]; o.y = packed[1]; o.z = packed[2]; o.w = packed[3];
  *(uint4*)(tw + base) = o;
}

// ---------------------------------------------------------------------------
// Kernel 4: int8 WMMA GEMM, async-to-LDS double-buffered B tile.
//   out[m,n] = (recip_s[m] * mean|w|) * sum_k q[m,k]*tw[n,k]
// Block: 256 threads = 8 waves. Block tile: 128 (M) x 128 (N). K step: 128.
// Wave tile: 16 (M) x 128 (N) -> 8 independent accumulator chains.
// LDS: 2 x (128 x 128) int8 B tiles = 32 KB; GLOBAL_LOAD_ASYNC_TO_LDS_B128
// staging (ASYNCcnt), one barrier per K step.
// Per K step per wave: 16x v_wmma_i32_16x16x64_iu8.
// B fragments preloaded per K-half into 8 distinct register sets so the
// ds_load -> wmma waits can be partial instead of lock-step dscnt==0.
// ---------------------------------------------------------------------------
__global__ __launch_bounds__(256) void bitlinear_gemm_kernel(
    const int8_t* __restrict__ qa, const int8_t* __restrict__ tw,
    const float* __restrict__ recip_s, const float* __restrict__ scal,
    float* __restrict__ out) {
  __shared__ __align__(16) uint8_t ldsB[2][128 * 128];

  const int tid  = threadIdx.x;
  const int lane = tid & 31;
  const int wave = tid >> 5;
  const int hi   = lane >> 4;   // 0 or 1
  const int nl   = lane & 15;

  const int nBlk  = blockIdx.x * 128;
  const int mWave = blockIdx.y * 128 + wave * 16;

  // per-lane A row pointer (ISA 8-bit A layout: hi lanes shifted +8 bytes)
  const uint8_t* aRow = (const uint8_t*)qa + (size_t)(mWave + nl) * KDIM + (hi << 3);

  // B staging: thread t stages 64 bytes of tile row (t>>1), byte chunk (t&1)*64
  const int bRow = tid >> 1;           // 0..127 (column n within tile)
  const int bOff = (tid & 1) << 6;     // 0, 64
  const uint8_t* bSrc = (const uint8_t*)tw + (size_t)(nBlk + bRow) * KDIM + bOff;
  const int ldsOff = bRow * 128 + bOff;

  // LDS byte addresses (low 32 bits of the generic pointer == LDS offset)
  const uint32_t ldsAddr0 = (uint32_t)(uintptr_t)&ldsB[0][ldsOff];
  const uint32_t ldsAddr1 = (uint32_t)(uintptr_t)&ldsB[1][ldsOff];

  // async stage 64 bytes: instruction offset applies to BOTH the global and
  // the LDS address (ISA 10.7)
#define STAGE_ASYNC(LADDR, GSRC)                                            \
  {                                                                         \
    uint64_t ga_ = (uint64_t)(uintptr_t)(GSRC);                             \
    asm volatile(                                                           \
        "global_load_async_to_lds_b128 %0, %1, off\n\t"                     \
        "global_load_async_to_lds_b128 %0, %1, off offset:16\n\t"           \
        "global_load_async_to_lds_b128 %0, %1, off offset:32\n\t"           \
        "global_load_async_to_lds_b128 %0, %1, off offset:48"               \
        ::"v"(LADDR), "v"(ga_)                                              \
        : "memory");                                                        \
  }
#define ASYNC_WAIT() asm volatile("s_wait_asynccnt 0x0" ::: "memory")

  // prologue: stage K-tile 0 into buffer 0
  STAGE_ASYNC(ldsAddr0, bSrc);

  v8i acc[8] = {};

#pragma clang loop unroll(disable)
  for (int k = 0; k < KDIM; k += 128) {
    const int cur = (k >> 7) & 1;

    ASYNC_WAIT();      // this wave's portion of buf[cur] is in LDS
    __syncthreads();   // all waves' portions are in LDS (and all prior reads done)

    // kick off next tile into the alternate buffer; overlapped with compute
    if (k + 128 < KDIM) {
      STAGE_ASYNC(cur ? ldsAddr0 : ldsAddr1, bSrc + k + 128);
      __builtin_prefetch(aRow + k + 128, 0, 1);  // global_prefetch_b8 next A
    }

    // A fragments for BOTH 64-wide K halves, loaded up front (8x b64, one clause)
    union { v8i v[2]; uint64_t u[8]; } A;
    {
      const uint64_t* ap = (const uint64_t*)(aRow + k);
      A.u[0] = ap[0];   // h=0: K = kb +  0.. 7
      A.u[1] = ap[2];   //      K = kb + 16..23
      A.u[2] = ap[4];   //      K = kb + 32..39
      A.u[3] = ap[6];   //      K = kb + 48..55
      A.u[4] = ap[8];   // h=1: K = 64 + kb +  0.. 7
      A.u[5] = ap[10];  //      ...
      A.u[6] = ap[12];
      A.u[7] = ap[14];
    }

#pragma unroll
    for (int h = 0; h < 2; ++h) {
      // preload all 8 B fragments for this K-half into distinct registers
      union { v8i v; uint64_t u[4]; } B[8];
#pragma unroll
      for (int j = 0; j < 8; ++j) {
        const uint64_t* bp =
            (const uint64_t*)&ldsB[cur][(j * 16 + nl) * 128 + h * 64 + (hi << 3)];
        B[j].u[0] = bp[0];
        B[j].u[1] = bp[2];
        B[j].u[2] = bp[4];
        B[j].u[3] = bp[6];
      }
      // 8 independent WMMAs; same-accumulator ops are 8 apart
#pragma unroll
      for (int j = 0; j < 8; ++j) {
        acc[j] = __builtin_amdgcn_wmma_i32_16x16x64_iu8(
            true, A.v[h], true, B[j].v, acc[j], false, false);
      }
    }
  }

  // epilogue: C/D layout — VGPR r holds M = r + 8*hi, N = lane&15
  const float invw = scal[1];
#pragma unroll
  for (int j = 0; j < 8; ++j) {
    const int n = nBlk + j * 16 + nl;
#pragma unroll
    for (int r = 0; r < 8; ++r) {
      const int m = mWave + hi * 8 + r;
      out[(size_t)m * NDIM + n] = (float)acc[j][r] * (recip_s[m] * invw);
    }
  }
#undef STAGE_ASYNC
#undef ASYNC_WAIT
}

// ---------------------------------------------------------------------------
// Host-side launcher
// ---------------------------------------------------------------------------
extern "C" void kernel_launch(void* const* d_in, const int* in_sizes, int n_in,
                              void* d_out, int out_size, void* d_ws, size_t ws_size,
                              hipStream_t stream) {
  const float* x = (const float*)d_in[0];   // [4, 2048, 4096] fp32
  const float* w = (const float*)d_in[1];   // [4096, 4096] fp32
  float* out = (float*)d_out;               // [4, 2048, 4096] fp32

  uint8_t* ws = (uint8_t*)d_ws;
  const size_t OFF_Q    = 0;                                 // 32 MB  int8 q
  const size_t OFF_TW   = OFF_Q  + (size_t)MDIM * KDIM;      // 16 MB  int8 tw
  const size_t OFF_RS   = OFF_TW + (size_t)NDIM * KDIM;      // 32 KB  recip_s
  const size_t OFF_PART = OFF_RS + (size_t)MDIM * 4;         // 16 KB  row partials
  const size_t OFF_SCAL = OFF_PART + (size_t)NDIM * 4;       // 8 B    {scale_w, 1/scale_w}

  int8_t* q       = (int8_t*)(ws + OFF_Q);
  int8_t* tw      = (int8_t*)(ws + OFF_TW);
  float*  recip_s = (float*)(ws + OFF_RS);
  float*  part    = (float*)(ws + OFF_PART);
  float*  scal    = (float*)(ws + OFF_SCAL);

  quant_act_kernel<<<MDIM, 256, 0, stream>>>(x, q, recip_s);
  wabs_partial_kernel<<<NDIM, 256, 0, stream>>>(w, part);
  wabs_final_kernel<<<1, 256, 0, stream>>>(part, scal);
  ternarize_kernel<<<(int)(((size_t)NDIM * KDIM) / (256 * 16)), 256, 0, stream>>>(w, scal, tw);

  dim3 grid(NDIM / 128, MDIM / 128);
  bitlinear_gemm_kernel<<<grid, 256, 0, stream>>>(q, tw, recip_s, scal, out);
}
